// LuangAttention_31671088840920
// MI455X (gfx1250) — matrix operand order repaired and checked by reading, not compiled, not run on hardware
//
#include <hip/hip_runtime.h>
#include <hip/hip_bf16.h>

// Luong 'general' attention for MI455X (gfx1250, wave32, WMMA).
// B=16, TQ=1024, TK=4096, D=512. All GEMMs on v_wmma_f32_16x16x32_bf16.
// GEMM inner loops are pure {global_load_b128 -> v_wmma}: all fp32->bf16
// conversion is hoisted into streaming convert/transpose kernels (or done
// once per workgroup), so no WMMA->VALU WAR hazards (v_nop groups) remain.

typedef __attribute__((ext_vector_type(16))) __bf16 v16bf;
typedef __attribute__((ext_vector_type(8)))  __bf16 v8bf;
typedef __attribute__((ext_vector_type(8)))  float  v8f;

#define B_  16
#define TQ_ 1024
#define TK_ 4096
#define D_  512

// ---------------------------------------------------------------------------
// A-operand load: 16x32 bf16 tile, bf16 row-major source (row stride = ld).
// Layout (matches B operand): lane half h holds K = h*16+i (i<16)
// -> two contiguous 16-byte chunks -> 2x b128.
// ---------------------------------------------------------------------------
__device__ __forceinline__ v16bf load_a_bf16(const __bf16* __restrict__ rowk0,
                                             int half_) {
  v8bf lo = *(const v8bf*)(rowk0 + half_ * 16);
  v8bf hi = *(const v8bf*)(rowk0 + half_ * 16 + 8);
  return __builtin_shufflevector(lo, hi, 0, 1, 2, 3, 4, 5, 6, 7,
                                 8, 9, 10, 11, 12, 13, 14, 15);
}

// Same gather from an fp32 source (used once per workgroup in score kernel).
__device__ __forceinline__ v16bf load_a_from_f32(const float* __restrict__ rowk0,
                                                 int half_) {
  v16bf a;
#pragma unroll
  for (int p = 0; p < 8; ++p) {
    int kp = half_ * 16 + 2 * p;
    float2 f = *(const float2*)(rowk0 + kp);
    a[2 * p]     = (__bf16)f.x;
    a[2 * p + 1] = (__bf16)f.y;
  }
  return a;
}

// ---------------------------------------------------------------------------
// Streaming fp32 -> bf16 convert (8 elements / thread).
// ---------------------------------------------------------------------------
__global__ __launch_bounds__(256)
void convert_f32_to_bf16(const float* __restrict__ src, __bf16* __restrict__ dst) {
  size_t i = ((size_t)blockIdx.x * 256 + threadIdx.x) * 8;
  float4 f0 = *(const float4*)(src + i);
  float4 f1 = *(const float4*)(src + i + 4);
  v8bf o;
  o[0] = (__bf16)f0.x; o[1] = (__bf16)f0.y; o[2] = (__bf16)f0.z; o[3] = (__bf16)f0.w;
  o[4] = (__bf16)f1.x; o[5] = (__bf16)f1.y; o[6] = (__bf16)f1.z; o[7] = (__bf16)f1.w;
  *(v8bf*)(dst + i) = o;
}

// ---------------------------------------------------------------------------
// Tiled transpose + fp32->bf16 convert:  dst[C][R] = (bf16) src[R][C]
// (batched over gridDim.z). 32x32 tiles through LDS, block (32,8).
// ---------------------------------------------------------------------------
__global__ __launch_bounds__(256)
void transpose_to_bf16(const float* __restrict__ src, __bf16* __restrict__ dst,
                       int R, int C) {
  __shared__ float tile[32][33];
  src += (size_t)blockIdx.z * R * C;
  dst += (size_t)blockIdx.z * R * C;
  int r0 = blockIdx.x * 32, c0 = blockIdx.y * 32;
  int tx = threadIdx.x, ty = threadIdx.y;
#pragma unroll
  for (int j = 0; j < 32; j += 8)
    tile[ty + j][tx] = src[(size_t)(r0 + ty + j) * C + (c0 + tx)];
  __syncthreads();
#pragma unroll
  for (int j = 0; j < 32; j += 8)
    dst[(size_t)(c0 + ty + j) * R + (r0 + tx)] = (__bf16)tile[tx][ty + j];
}

// ---------------------------------------------------------------------------
// proj[b][tk][e] = bf16( sum_d enc[b][tk][d] * W[d][e] + bias[e] )
// One wave = one 16x32 output tile (two accumulator chains sharing A).
// Inner loop: 6x global_load_b128 + 2x v_wmma per K-step, no VALU.
// ---------------------------------------------------------------------------
__global__ __launch_bounds__(128)
void proj_gemm_kernel(const __bf16* __restrict__ encb, const __bf16* __restrict__ Wt,
                      const float* __restrict__ bias, __bf16* __restrict__ proj) {
  int wid   = blockIdx.x * 4 + (threadIdx.x >> 5);
  int lane  = threadIdx.x & 31;
  int half_ = lane >> 4, lh = lane & 15;
  int b   = wid >> 12;               // (TK/16)*(D/32) = 4096 tile-pairs / batch
  int rem = wid & 4095;
  int tk0 = (rem >> 4) * 16;         // 256 tk tiles
  int e0  = (rem & 15) * 32;         // 16 e tile-pairs

  const __bf16* arow  = encb + ((size_t)b * TK_ + tk0 + lh) * D_;
  const __bf16* brow0 = Wt + (size_t)(e0 + lh) * D_ + half_ * 16;
  const __bf16* brow1 = brow0 + (size_t)16 * D_;

  float bv0 = bias[e0 + lh];
  float bv1 = bias[e0 + 16 + lh];
  v8f c0, c1;
#pragma unroll
  for (int r = 0; r < 8; ++r) { c0[r] = bv0; c1[r] = bv1; }

#pragma unroll
  for (int ks = 0; ks < 16; ++ks) {
    v16bf a   = load_a_bf16(arow + ks * 32, half_);
    v16bf bm0 = *(const v16bf*)(brow0 + ks * 32);
    v16bf bm1 = *(const v16bf*)(brow1 + ks * 32);
    c0 = __builtin_amdgcn_wmma_f32_16x16x32_bf16(false, a, false, bm0,
                                                 (short)0, c0, false, false);
    c1 = __builtin_amdgcn_wmma_f32_16x16x32_bf16(false, a, false, bm1,
                                                 (short)0, c1, false, false);
  }

  // C/D layout: N = lane&15, M = (lane>>4)*8 + r
  __bf16* out = proj + ((size_t)b * TK_ + tk0 + half_ * 8) * D_ + e0 + lh;
#pragma unroll
  for (int r = 0; r < 8; ++r) {
    out[(size_t)r * D_]      = (__bf16)c0[r];
    out[(size_t)r * D_ + 16] = (__bf16)c1[r];
  }
}

// ---------------------------------------------------------------------------
// Fused score + softmax. One 256-thread (8-wave) workgroup handles 16 queries
// against all TK=4096 keys; the full 16x4096 fp32 score block (256 KB) lives
// in LDS (dynamic alloc; CDNA5 WGP has 320 KB). Wave w covers key columns
// [w*512, (w+1)*512) as 16 n-tile pairs; dec A-tile is register-resident
// (16 K-steps x 8 VGPRs, converted once) and reused for all 32 column tiles.
// Emits fp32 probabilities to d_out AND bf16 probabilities for the context
// GEMM's A operand.
// ---------------------------------------------------------------------------
__global__ __launch_bounds__(256)
void score_softmax_kernel(const float* __restrict__ dec,
                          const __bf16* __restrict__ proj,
                          float* __restrict__ align_out,
                          __bf16* __restrict__ p_bf16) {
  extern __shared__ float s[];       // [16][TK_] fp32 scores
  int b  = blockIdx.x >> 6;          // TQ/16 = 64 q-tiles per batch
  int q0 = (blockIdx.x & 63) * 16;
  int w     = threadIdx.x >> 5;      // wave 0..7
  int lane  = threadIdx.x & 31;
  int half_ = lane >> 4, lh = lane & 15;

  const float* arow = dec + ((size_t)b * TQ_ + q0 + lh) * D_;
  v16bf areg[16];
#pragma unroll
  for (int ks = 0; ks < 16; ++ks) areg[ks] = load_a_from_f32(arow + ks * 32, half_);

  // B[K=d][N=tk] = proj[tk][d]: contiguous in d per lane.
  const __bf16* pbase = proj + (size_t)b * TK_ * D_ + half_ * 16;

  for (int nt = 0; nt < 16; ++nt) {
    int n0 = w * 512 + nt * 32;
    const __bf16* prow0 = pbase + (size_t)(n0 + lh) * D_;
    const __bf16* prow1 = prow0 + (size_t)16 * D_;
    v8f c0 = {}, c1 = {};
#pragma unroll
    for (int ks = 0; ks < 16; ++ks) {
      v16bf bm0 = *(const v16bf*)(prow0 + ks * 32);
      v16bf bm1 = *(const v16bf*)(prow1 + ks * 32);
      c0 = __builtin_amdgcn_wmma_f32_16x16x32_bf16(false, areg[ks], false, bm0,
                                                   (short)0, c0, false, false);
      c1 = __builtin_amdgcn_wmma_f32_16x16x32_bf16(false, areg[ks], false, bm1,
                                                   (short)0, c1, false, false);
    }
#pragma unroll
    for (int r = 0; r < 8; ++r) {
      s[(half_ * 8 + r) * TK_ + n0 + lh]      = c0[r];
      s[(half_ * 8 + r) * TK_ + n0 + 16 + lh] = c1[r];
    }
  }
  __syncthreads();

  // Softmax: 16 threads per row, 16 rows; 16-wide shuffle reductions.
  int r = threadIdx.x >> 4;
  int t = threadIdx.x & 15;
  float* row = s + r * TK_;

  float m = -3.402823466e38f;
  for (int c0 = t; c0 < TK_; c0 += 16) m = fmaxf(m, row[c0]);
#pragma unroll
  for (int o = 8; o > 0; o >>= 1) m = fmaxf(m, __shfl_xor(m, o, 16));

  float sum = 0.f;
  for (int c0 = t; c0 < TK_; c0 += 16) {
    float e = __expf(row[c0] - m);
    row[c0] = e;
    sum += e;
  }
#pragma unroll
  for (int o = 8; o > 0; o >>= 1) sum += __shfl_xor(sum, o, 16);
  float inv = 1.0f / sum;

  float*  orow = align_out + ((size_t)b * TQ_ + q0 + r) * TK_;
  __bf16* prow = p_bf16   + ((size_t)b * TQ_ + q0 + r) * TK_;
  for (int c0 = t; c0 < TK_; c0 += 16) {
    float p = row[c0] * inv;
    orow[c0] = p;
    prow[c0] = (__bf16)p;
  }
}

// ---------------------------------------------------------------------------
// context[b][q][e] = sum_k P[b][q][k] * enc[b][k][e]
// A = P_bf16 tile (2x b128/step), B = encT[b][e][k] (bf16, contiguous K).
// Inner loop: 6x global_load_b128 + 2x v_wmma per K-step, 128 steps, no VALU.
// ---------------------------------------------------------------------------
__global__ __launch_bounds__(128)
void context_gemm_kernel(const __bf16* __restrict__ p_bf16,
                         const __bf16* __restrict__ encT,
                         float* __restrict__ ctx) {
  int wid   = blockIdx.x * 4 + (threadIdx.x >> 5);
  int lane  = threadIdx.x & 31;
  int half_ = lane >> 4, lh = lane & 15;
  int b   = wid >> 10;               // (TQ/16)*(D/32) = 1024 tile-pairs / batch
  int rem = wid & 1023;
  int q0  = (rem >> 4) * 16;
  int e0  = (rem & 15) * 32;

  const __bf16* arow  = p_bf16 + ((size_t)b * TQ_ + q0 + lh) * TK_;
  const __bf16* brow0 = encT + ((size_t)b * D_ + e0 + lh) * TK_ + half_ * 16;
  const __bf16* brow1 = brow0 + (size_t)16 * TK_;

  v8f c0 = {}, c1 = {};
#pragma unroll 8
  for (int ks = 0; ks < 128; ++ks) {
    v16bf a   = load_a_bf16(arow + ks * 32, half_);
    v16bf bm0 = *(const v16bf*)(brow0 + ks * 32);
    v16bf bm1 = *(const v16bf*)(brow1 + ks * 32);
    c0 = __builtin_amdgcn_wmma_f32_16x16x32_bf16(false, a, false, bm0,
                                                 (short)0, c0, false, false);
    c1 = __builtin_amdgcn_wmma_f32_16x16x32_bf16(false, a, false, bm1,
                                                 (short)0, c1, false, false);
  }

  float* out = ctx + ((size_t)b * TQ_ + q0 + half_ * 8) * D_ + e0 + lh;
#pragma unroll
  for (int r = 0; r < 8; ++r) {
    out[(size_t)r * D_]      = c0[r];
    out[(size_t)r * D_ + 16] = c1[r];
  }
}

// ---------------------------------------------------------------------------
extern "C" void kernel_launch(void* const* d_in, const int* in_sizes, int n_in,
                              void* d_out, int out_size, void* d_ws, size_t ws_size,
                              hipStream_t stream) {
  const float* dec  = (const float*)d_in[0];   // [B, TQ, D]
  const float* enc  = (const float*)d_in[1];   // [B, TK, D]
  const float* W    = (const float*)d_in[2];   // [D, D]
  const float* bias = (const float*)d_in[3];   // [D]

  // Workspace (bf16): Wt [D][D] | enc_bf16 [B][TK][D] | encT [B][D][TK]
  //                  | proj [B][TK][D] | P [B][TQ][TK]
  const size_t szWt   = (size_t)D_ * D_ * 2;           // 512 KB
  const size_t szEnc  = (size_t)B_ * TK_ * D_ * 2;     // 64 MB
  const size_t szP    = (size_t)B_ * TQ_ * TK_ * 2;    // 128 MB
  char* ws = (char*)d_ws;
  __bf16* Wt    = (__bf16*)ws;
  __bf16* encb  = (__bf16*)(ws + szWt);
  __bf16* encT  = (__bf16*)(ws + szWt + szEnc);
  __bf16* proj  = (__bf16*)(ws + szWt + 2 * szEnc);
  __bf16* Pbf   = (__bf16*)(ws + szWt + 3 * szEnc);
  (void)szP; (void)ws_size;

  float* ctx   = (float*)d_out;                          // [B, TQ, D]
  float* align = (float*)d_out + (size_t)B_ * TQ_ * D_;  // [B, TQ, TK]

  dim3 tb(32, 8);
  // 1) enc_bf16 = bf16(enc)   (streaming, 8 elems/thread)
  convert_f32_to_bf16<<<(B_ * TK_ * D_) / (256 * 8), 256, 0, stream>>>(enc, encb);
  // 2) Wt[e][d] = bf16(W[d][e])
  transpose_to_bf16<<<dim3(D_ / 32, D_ / 32, 1), tb, 0, stream>>>(W, Wt, D_, D_);
  // 3) encT[b][e][k] = bf16(enc[b][k][e])
  transpose_to_bf16<<<dim3(TK_ / 32, D_ / 32, B_), tb, 0, stream>>>(enc, encT, TK_, D_);
  // 4) proj = enc @ W + b   (bf16 out)
  proj_gemm_kernel<<<(B_ * (TK_ / 16) * (D_ / 32)) / 4, 128, 0, stream>>>(encb, Wt, bias, proj);
  // 5) fused score + softmax -> fp32 probs (d_out) + bf16 probs (ws)
  score_softmax_kernel<<<B_ * (TQ_ / 16), 256, 16 * TK_ * sizeof(float), stream>>>(dec, proj, align, Pbf);
  // 6) context = P @ enc
  context_gemm_kernel<<<(B_ * (TQ_ / 16) * (D_ / 32)) / 4, 128, 0, stream>>>(Pbf, encT, ctx);
}
